// Model_23192823398863
// MI455X (gfx1250) — matrix operand order repaired and checked by reading, not compiled, not run on hardware
//
#include <hip/hip_runtime.h>
#include <math.h>

// ---------------------------------------------------------------------------
// MI455X (gfx1250) implementation.
// All heavy math is routed through V_WMMA_F32_16X16X32_F16 (wave32, f32
// accumulate). Each wave register-blocks a 32x32 output tile (2x2 WMMA tiles)
// so every loaded A/B fragment feeds two WMMAs. The complex GEMM uses three
// accumulator sets (P=ArWr, M=AiWi, I=ArWi+AiWr; real = P-M in the epilogue)
// so the k-loop contains no VALU ops -> no VALU->WMMA hazard NOPs.
// FFT/iRFFT are expressed as GEMMs against generated DFT matrices. F=97 is
// padded to 128 (k-dim must be a multiple of 32 for the 16x16x32 WMMA loop).
// ---------------------------------------------------------------------------

typedef __attribute__((ext_vector_type(16))) _Float16 v16h;
typedef __attribute__((ext_vector_type(8)))  _Float16 v8h;
typedef __attribute__((ext_vector_type(8)))  float    v8f;
typedef _Float16 half_t;

namespace dims {
constexpr int B = 8, C = 3, N = 128, M = 8, D = 512, H = 8, HD = 64;
constexpr int S = 96, F = 97, L = 192, P = 96, FP = 128;
constexpr int NBC = B * C;          // 24
constexpr int RQ  = NBC * N;        // 3072  query rows (b,c,n)
constexpr int RKV = RQ * M;         // 24576 kv rows (b,c,n,m)
constexpr int KV  = N * M;          // 1024  kv tokens per (b,c)
}

__constant__ float kTwoPi = 6.28318530717958647692f;

// ---------------- WMMA helpers (ISA 7.12.2 layouts) -------------------------

__device__ __forceinline__ v8f wmma16(v16h a, v16h b, v8f c) {
  // (neg_a, A, neg_b, B, c_mod, C, reuse_a, reuse_b)
  return __builtin_amdgcn_wmma_f32_16x16x32_f16(false, a, false, b,
                                                (short)0, c, false, false);
}

// A 16x32 f16: lane<16 holds row m=lane, halves 0..7 = K[k..k+7],
// halves 8..15 = K[k+16..k+23]; lanes>=16 same row, K offset +8.
__device__ __forceinline__ v16h load_a16(const half_t* __restrict__ p) {
  v8h lo = *(const v8h*)(p);
  v8h hi = *(const v8h*)(p + 16);
  v16h r;
#pragma unroll
  for (int j = 0; j < 8; ++j) { r[j] = lo[j]; r[j + 8] = hi[j]; }
  return r;
}

// B 32x16 f16: lane<16 = column n=lane, halves 0..7 = K[k..k+7],
// halves 8..15 = K[k+16..k+23]; lanes>=16 same column, K offset +8.
// B-col n == weight row n (we compute x @ W^T with row-major W).
__device__ __forceinline__ v16h load_b16(const half_t* __restrict__ p) {
  v8h lo = *(const v8h*)(p);
  v8h hi = *(const v8h*)(p + 16);
  v16h r;
#pragma unroll
  for (int j = 0; j < 8; ++j) { r[j] = lo[j]; r[j + 8] = hi[j]; }
  return r;
}

// ---------------- real GEMM: C = act(A1 W1^T [+ A2 W2^T] + bias [+ res]) ----
// 32x32 tile per wave (2x2 WMMA register blocking). Batched via z =
// blockIdx.z decomposed as zo=z/zDiv, zi=z%zDiv with independent outer/inner
// strides per operand (covers the per-(b,c,h) attention GEMMs).

template <int DUAL>
__global__ __launch_bounds__(128) void gemm_rr_kernel(
    const half_t* __restrict__ A1, long lda1, long aOut, long aIn,
    const half_t* __restrict__ W1, long ldw1, long wOut, long wIn,
    const half_t* __restrict__ A2, long lda2,
    const half_t* __restrict__ W2, long ldw2,
    const float* __restrict__ bias,
    const half_t* __restrict__ resid, long ldr,
    half_t* __restrict__ Ch, float* __restrict__ Cf,
    long ldc, long cOut, long cIn,
    int Mdim, int Ndim, int Kdim, int zDiv, int doRelu) {
  const int lane = threadIdx.x & 31;
  const int wave = threadIdx.x >> 5;
  const int m0 = (blockIdx.y * 4 + wave) * 32;
  const int n0 = blockIdx.x * 32;
  if (m0 >= Mdim) return;  // wave-uniform (EXEC stays all-ones for WMMA)

  const int z = blockIdx.z;
  const int zo = z / zDiv, zi = z % zDiv;
  const long aOff = (long)zo * aOut + (long)zi * aIn;
  const long wOff = (long)zo * wOut + (long)zi * wIn;
  const long cOff = (long)zo * cOut + (long)zi * cIn;

  const int r15 = lane & 15;
  const int alo = (lane < 16) ? 0 : 8;
  const int blo = (lane < 16) ? 0 : 8;

  const half_t* a1 = A1 + aOff + (long)(m0 + r15) * lda1 + alo;
  const half_t* w1 = W1 + wOff + (long)(n0 + r15) * ldw1 + blo;
  const half_t* a2 = DUAL ? (A2 + aOff + (long)(m0 + r15) * lda2 + alo) : nullptr;
  const half_t* w2 = DUAL ? (W2 + wOff + (long)(n0 + r15) * ldw2 + blo) : nullptr;

  v8f acc[2][2] = {};
  for (int k = 0; k < Kdim; k += 32) {
    v16h av[2], bv[2];
    av[0] = load_a16(a1 + k);
    av[1] = load_a16(a1 + 16 * lda1 + k);
    bv[0] = load_b16(w1 + k);
    bv[1] = load_b16(w1 + 16 * ldw1 + k);
#pragma unroll
    for (int mi = 0; mi < 2; ++mi)
#pragma unroll
      for (int ni = 0; ni < 2; ++ni)
        acc[mi][ni] = wmma16(av[mi], bv[ni], acc[mi][ni]);
    if (DUAL) {
      v16h av2[2], bv2[2];
      av2[0] = load_a16(a2 + k);
      av2[1] = load_a16(a2 + 16 * lda2 + k);
      bv2[0] = load_b16(w2 + k);
      bv2[1] = load_b16(w2 + 16 * ldw2 + k);
#pragma unroll
      for (int mi = 0; mi < 2; ++mi)
#pragma unroll
        for (int ni = 0; ni < 2; ++ni)
          acc[mi][ni] = wmma16(av2[mi], bv2[ni], acc[mi][ni]);
    }
  }

#pragma unroll
  for (int mi = 0; mi < 2; ++mi) {
    const int mb = m0 + mi * 16 + ((lane < 16) ? 0 : 8);
#pragma unroll
    for (int ni = 0; ni < 2; ++ni) {
      const int col = n0 + ni * 16 + r15;
      const float bv = bias ? bias[col] : 0.0f;
#pragma unroll
      for (int r = 0; r < 8; ++r) {
        float v = acc[mi][ni][r] + bv;
        if (resid) v += (float)resid[(long)(mb + r) * ldr + col];
        if (doRelu) v = fmaxf(v, 0.0f);
        const long idx = cOff + (long)(mb + r) * ldc + col;
        if (Cf) Cf[idx] = v; else Ch[idx] = (half_t)v;
      }
    }
  }
}

// ---------------- complex GEMM: (Cr+iCi) = act((Ar+iAi)(Wr+iWi)^T + b) ------
// 32x32 tile per wave, 3 accumulator sets (P/M/I); real = P - M folded into
// the epilogue so the k-loop is pure loads + 16 WMMAs (no VALU hazards).

__global__ __launch_bounds__(128) void gemm_cc_kernel(
    const half_t* __restrict__ Ar, const half_t* __restrict__ Ai, long lda,
    const half_t* __restrict__ Wr, const half_t* __restrict__ Wi, long ldw,
    const float* __restrict__ br, const float* __restrict__ bi,
    half_t* __restrict__ Cr, half_t* __restrict__ Ci, long ldc,
    int Mdim, int Ndim, int Kdim, int crelu) {
  const int lane = threadIdx.x & 31;
  const int wave = threadIdx.x >> 5;
  const int m0 = (blockIdx.y * 4 + wave) * 32;
  const int n0 = blockIdx.x * 32;
  if (m0 >= Mdim) return;

  const int r15 = lane & 15;
  const int alo = (lane < 16) ? 0 : 8;
  const int blo = (lane < 16) ? 0 : 8;

  const half_t* ar = Ar + (long)(m0 + r15) * lda + alo;
  const half_t* ai = Ai + (long)(m0 + r15) * lda + alo;
  const half_t* wr = Wr + (long)(n0 + r15) * ldw + blo;
  const half_t* wi = Wi + (long)(n0 + r15) * ldw + blo;

  v8f accP[2][2] = {};
  v8f accM[2][2] = {};
  v8f accI[2][2] = {};
  for (int k = 0; k < Kdim; k += 32) {
    v16h arv[2], aiv[2], wrv[2], wiv[2];
    arv[0] = load_a16(ar + k);
    arv[1] = load_a16(ar + 16 * lda + k);
    aiv[0] = load_a16(ai + k);
    aiv[1] = load_a16(ai + 16 * lda + k);
    wrv[0] = load_b16(wr + k);
    wrv[1] = load_b16(wr + 16 * ldw + k);
    wiv[0] = load_b16(wi + k);
    wiv[1] = load_b16(wi + 16 * ldw + k);
#pragma unroll
    for (int mi = 0; mi < 2; ++mi)
#pragma unroll
      for (int ni = 0; ni < 2; ++ni) {
        accP[mi][ni] = wmma16(arv[mi], wrv[ni], accP[mi][ni]);
        accM[mi][ni] = wmma16(aiv[mi], wiv[ni], accM[mi][ni]);
        accI[mi][ni] = wmma16(arv[mi], wiv[ni], accI[mi][ni]);
        accI[mi][ni] = wmma16(aiv[mi], wrv[ni], accI[mi][ni]);
      }
  }

#pragma unroll
  for (int mi = 0; mi < 2; ++mi) {
    const int mb = m0 + mi * 16 + ((lane < 16) ? 0 : 8);
#pragma unroll
    for (int ni = 0; ni < 2; ++ni) {
      const int col = n0 + ni * 16 + r15;
      const float bvr = br ? br[col] : 0.0f;
      const float bvi = bi ? bi[col] : 0.0f;
#pragma unroll
      for (int r = 0; r < 8; ++r) {
        float vr = accP[mi][ni][r] - accM[mi][ni][r] + bvr;
        float vi = accI[mi][ni][r] + bvi;
        if (crelu) { vr = fmaxf(vr, 0.0f); vi = fmaxf(vi, 0.0f); }
        const long idx = (long)(mb + r) * ldc + col;
        Cr[idx] = (half_t)vr;
        Ci[idx] = (half_t)vi;
      }
    }
  }
}

// ---------------- small support kernels -------------------------------------

__global__ void cvt_pad_kernel(const float* __restrict__ src,
                               half_t* __restrict__ dst,
                               int Rs, int Cs, int Rd, int Cd) {
  long idx = (long)blockIdx.x * blockDim.x + threadIdx.x;
  if (idx >= (long)Rd * Cd) return;
  int r = (int)(idx / Cd), c = (int)(idx % Cd);
  float v = (r < Rs && c < Cs) ? src[(long)r * Cs + c] : 0.0f;
  dst[idx] = (half_t)v;
}

__global__ void pad_bias_kernel(const float* __restrict__ s,
                                float* __restrict__ d, int Cs, int Cd) {
  int i = blockIdx.x * blockDim.x + threadIdx.x;
  if (i < Cd) d[i] = (i < Cs) ? s[i] : 0.0f;
}

// rfft DFT matrices: Wc[f][t]=cos(2pi f t/L), Ws[f][t]=-sin(2pi f t/L), f<F
__global__ void gen_fft_kernel(half_t* __restrict__ Wc, half_t* __restrict__ Ws) {
  int idx = blockIdx.x * blockDim.x + threadIdx.x;
  if (idx >= dims::FP * dims::L) return;
  int f = idx / dims::L, t = idx % dims::L;
  float c = 0.f, s = 0.f;
  if (f < dims::F) {
    float th = kTwoPi * (float)f * (float)t / (float)dims::L;
    c = cosf(th);
    s = -sinf(th);
  }
  Wc[idx] = (half_t)c;
  Ws[idx] = (half_t)s;
}

// irfft: x[t] = sum_f wf*(Xr cos - Xi sin)/L ; wf = 1 for f in {0,S}, else 2
__global__ void gen_ir_kernel(half_t* __restrict__ Cr, half_t* __restrict__ Ci) {
  int idx = blockIdx.x * blockDim.x + threadIdx.x;
  if (idx >= dims::L * dims::FP) return;
  int t = idx / dims::FP, f = idx % dims::FP;
  float cr = 0.f, ci = 0.f;
  if (f < dims::F) {
    float w = ((f == 0) || (f == dims::S)) ? 1.0f : 2.0f;
    w /= (float)dims::L;
    float th = kTwoPi * (float)f * (float)t / (float)dims::L;
    cr = w * cosf(th);
    ci = -w * sinf(th);
  }
  Cr[idx] = (half_t)cr;
  Ci[idx] = (half_t)ci;
}

// xf *= sigmoid(freq_weights[f])  (cols >= F are structurally zero)
__global__ void scale_xf_kernel(half_t* __restrict__ xr, half_t* __restrict__ xi,
                                const float* __restrict__ w) {
  long idx = (long)blockIdx.x * blockDim.x + threadIdx.x;
  if (idx >= (long)dims::RQ * dims::FP) return;
  int f = (int)(idx & (dims::FP - 1));
  if (f < dims::F) {
    float s = 1.0f / (1.0f + __expf(-w[f]));
    xr[idx] = (half_t)((float)xr[idx] * s);
    xi[idx] = (half_t)((float)xi[idx] * s);
  }
}

// xt[bcn,m,f] = xf[bcn,f] * temp[m,f]  (complex multiply)
__global__ void make_xt_kernel(const half_t* __restrict__ xr,
                               const half_t* __restrict__ xi,
                               const float* __restrict__ tr,
                               const float* __restrict__ ti,
                               half_t* __restrict__ xtr,
                               half_t* __restrict__ xti) {
  long idx = (long)blockIdx.x * blockDim.x + threadIdx.x;
  if (idx >= (long)dims::RKV * dims::FP) return;
  int f = (int)(idx & (dims::FP - 1));
  long row = idx >> 7;          // FP == 128
  int m = (int)(row & 7);
  long bcn = row >> 3;
  float a = (float)xr[bcn * dims::FP + f];
  float b = (float)xi[bcn * dims::FP + f];
  float c = 0.f, d = 0.f;
  if (f < dims::F) {
    c = tr[m * dims::F + f];
    d = ti[m * dims::F + f];
  }
  xtr[idx] = (half_t)(a * c - b * d);
  xti[idx] = (half_t)(a * d + b * c);
}

// per-(b,c,h) head transpose: Vt[z][e][q] = V[bc*KV+q][h*HD+e]
__global__ void transpose_v_kernel(const half_t* __restrict__ V,
                                   half_t* __restrict__ Vt) {
  long idx = (long)blockIdx.x * blockDim.x + threadIdx.x;
  const long total = (long)dims::NBC * dims::H * dims::HD * dims::KV;
  if (idx >= total) return;
  int q = (int)(idx & (dims::KV - 1));
  int e = (int)((idx >> 10) & (dims::HD - 1));
  long z = idx >> 16;           // / (KV*HD)
  long bc = z >> 3;
  int h = (int)(z & 7);
  Vt[idx] = V[(bc * dims::KV + q) * dims::D + h * dims::HD + e];
}

// wave-per-row softmax over 1024 kv logits (wave32 shfl reductions)
__global__ __launch_bounds__(256) void softmax_kernel(
    const float* __restrict__ S, half_t* __restrict__ Pout,
    int rows, int cols, float scale) {
  const int lane = threadIdx.x & 31;
  const int wave = threadIdx.x >> 5;
  const int row = blockIdx.x * 8 + wave;
  if (row >= rows) return;
  const float* sp = S + (long)row * cols;
  float mx = -3.0e38f;
  for (int j = lane; j < cols; j += 32) mx = fmaxf(mx, sp[j] * scale);
#pragma unroll
  for (int o = 16; o; o >>= 1) mx = fmaxf(mx, __shfl_xor(mx, o, 32));
  float sum = 0.f;
  for (int j = lane; j < cols; j += 32) sum += __expf(sp[j] * scale - mx);
#pragma unroll
  for (int o = 16; o; o >>= 1) sum += __shfl_xor(sum, o, 32);
  const float inv = 1.0f / sum;
  half_t* pp = Pout + (long)row * cols;
  for (int j = lane; j < cols; j += 32)
    pp[j] = (half_t)(__expf(sp[j] * scale - mx) * inv);
}

// ---------------------------------------------------------------------------

extern "C" void kernel_launch(void* const* d_in, const int* in_sizes, int n_in,
                              void* d_out, int out_size, void* d_ws,
                              size_t ws_size, hipStream_t stream) {
  (void)in_sizes; (void)n_in; (void)out_size; (void)ws_size;
  using namespace dims;

  // ---- input pointers (setup_inputs dict order, params flattened) ----
  const float* xp  = (const float*)d_in[0];
  const float* fw  = (const float*)d_in[1];
  const float* tmr = (const float*)d_in[2];
  const float* tmi = (const float*)d_in[3];
  const float* fc1wr = (const float*)d_in[4];
  const float* fc1wi = (const float*)d_in[5];
  const float* fc1br = (const float*)d_in[6];
  const float* fc1bi = (const float*)d_in[7];
  const float* fc2wr = (const float*)d_in[8];
  const float* fc2wi = (const float*)d_in[9];
  const float* fc2br = (const float*)d_in[10];
  const float* fc2bi = (const float*)d_in[11];
  const float* fc3wr = (const float*)d_in[12];
  const float* fc3wi = (const float*)d_in[13];
  const float* fc3br = (const float*)d_in[14];
  const float* fc3bi = (const float*)d_in[15];
  const float* fc4wr = (const float*)d_in[16];
  const float* fc4wi = (const float*)d_in[17];
  const float* fc4br = (const float*)d_in[18];
  const float* fc4bi = (const float*)d_in[19];
  const float* fc5wr = (const float*)d_in[20];
  const float* fc5wi = (const float*)d_in[21];
  const float* fc5br = (const float*)d_in[22];
  const float* fc5bi = (const float*)d_in[23];
  const float* AW[2][8];  // wq,bq,wk,bk,wv,bv,wo,bo  x {real, imag}
  for (int p2 = 0; p2 < 2; ++p2)
    for (int j = 0; j < 8; ++j)
      AW[p2][j] = (const float*)d_in[24 + p2 * 8 + j];
  const float* pw1 = (const float*)d_in[40];
  const float* pb1 = (const float*)d_in[41];
  const float* pw2 = (const float*)d_in[42];
  const float* pb2 = (const float*)d_in[43];

  // ---- workspace bump allocator (all offsets 256B aligned) ----
  char* baseP = (char*)d_ws;
  size_t off = 0;
  auto alloc = [&](size_t bytes) -> void* {
    void* p = baseP + off;
    off += (bytes + 255) & ~(size_t)255;
    return p;
  };
  auto ah = [&](size_t e) { return (half_t*)alloc(e * sizeof(half_t)); };
  auto af = [&](size_t e) { return (float*)alloc(e * sizeof(float)); };

  // f16 weights (padded to FP=128 on the F dimension)
  half_t* xp_h = ah((size_t)RQ * L);
  half_t* wfc  = ah((size_t)FP * L);
  half_t* wfs  = ah((size_t)FP * L);
  half_t* wir  = ah((size_t)L * FP);
  half_t* wii  = ah((size_t)L * FP);
  half_t* w1r = ah((size_t)D * FP); half_t* w1i = ah((size_t)D * FP);
  half_t* w2r = ah((size_t)D * D);  half_t* w2i = ah((size_t)D * D);
  half_t* w3r = ah((size_t)FP * D); half_t* w3i = ah((size_t)FP * D);
  half_t* w4r = ah((size_t)D * FP); half_t* w4i = ah((size_t)D * FP);
  half_t* w5r = ah((size_t)D * D);  half_t* w5i = ah((size_t)D * D);
  half_t* wAq[2], *wAk[2], *wAv[2], *wAo[2];
  for (int p2 = 0; p2 < 2; ++p2) {
    wAq[p2] = ah((size_t)D * D); wAk[p2] = ah((size_t)D * D);
    wAv[p2] = ah((size_t)D * D); wAo[p2] = ah((size_t)D * D);
  }
  half_t* wp1 = ah((size_t)(4 * P) * (2 * P));
  half_t* wp2 = ah((size_t)P * (4 * P));
  float* b3r = af(FP);
  float* b3i = af(FP);

  // activations (f16 unless noted)
  half_t* xfr = ah((size_t)RQ * FP);
  half_t* xfi = ah((size_t)RQ * FP);
  half_t* xtr = ah((size_t)RKV * FP);
  half_t* xti = ah((size_t)RKV * FP);
  half_t* t1r = ah((size_t)RKV * D);
  half_t* t1i = ah((size_t)RKV * D);
  half_t* er  = ah((size_t)RKV * D);
  half_t* ei  = ah((size_t)RKV * D);
  half_t* t2r = ah((size_t)RQ * D);
  half_t* t2i = ah((size_t)RQ * D);
  half_t* gr  = ah((size_t)RQ * D);
  half_t* gi  = ah((size_t)RQ * D);
  half_t* qb  = ah((size_t)RQ * D);
  half_t* kb  = t1r;  // alias: t1 dead after fc2
  half_t* vb  = t1i;  // alias
  half_t* vt  = ah((size_t)NBC * H * HD * KV);
  float*  sc  = af((size_t)NBC * H * N * KV);   // f32 scores for softmax
  half_t* pr  = ah((size_t)NBC * H * N * KV);   // f16 probs
  half_t* ao  = ah((size_t)RQ * D);
  half_t* outp[2] = { ah((size_t)RQ * D), ah((size_t)RQ * D) };
  half_t* o_r = ah((size_t)RQ * FP);
  half_t* o_i = ah((size_t)RQ * FP);
  half_t* ot  = ah((size_t)RQ * L);
  half_t* hb  = ah((size_t)RQ * 4 * P);

  // ---- launch helpers ----
  auto cvt = [&](const float* s, half_t* d, int Rs, int Cs, int Rd, int Cd) {
    long tot = (long)Rd * Cd;
    cvt_pad_kernel<<<dim3((unsigned)((tot + 255) / 256)), dim3(256), 0, stream>>>(
        s, d, Rs, Cs, Rd, Cd);
  };
  auto launch_rr = [&](const half_t* A, long lda, const half_t* W, long ldw,
                       const float* bias, const half_t* res, long ldr,
                       half_t* Ch, float* Cf, long ldc,
                       int Md, int Nd, int Kd, int relu) {
    dim3 g((unsigned)(Nd / 32), (unsigned)(Md / 128), 1);
    gemm_rr_kernel<0><<<g, dim3(128), 0, stream>>>(
        A, lda, 0, 0, W, ldw, 0, 0, nullptr, 0, nullptr, 0,
        bias, res, ldr, Ch, Cf, ldc, 0, 0, Md, Nd, Kd, 1, relu);
  };
  auto launch_rr_b = [&](const half_t* A, long lda, long aO, long aI,
                         const half_t* W, long ldw, long wO, long wI,
                         half_t* Ch, float* Cf, long ldc, long cO, long cI,
                         int Md, int Nd, int Kd, int Z, int zDiv) {
    unsigned gy = (unsigned)(Md / 128); if (gy == 0) gy = 1;
    dim3 g((unsigned)(Nd / 32), gy, (unsigned)Z);
    gemm_rr_kernel<0><<<g, dim3(128), 0, stream>>>(
        A, lda, aO, aI, W, ldw, wO, wI, nullptr, 0, nullptr, 0,
        nullptr, nullptr, 0, Ch, Cf, ldc, cO, cI, Md, Nd, Kd, zDiv, 0);
  };
  auto launch_cc = [&](const half_t* Ar, const half_t* Ai, long lda,
                       const half_t* Wr, const half_t* Wi, long ldw,
                       const float* br, const float* bi,
                       half_t* Cr2, half_t* Ci2, long ldc,
                       int Md, int Nd, int Kd, int crelu) {
    dim3 g((unsigned)(Nd / 32), (unsigned)(Md / 128), 1);
    gemm_cc_kernel<<<g, dim3(128), 0, stream>>>(
        Ar, Ai, lda, Wr, Wi, ldw, br, bi, Cr2, Ci2, ldc, Md, Nd, Kd, crelu);
  };

  // ---- weight preparation ----
  gen_fft_kernel<<<dim3((FP * L + 255) / 256), dim3(256), 0, stream>>>(wfc, wfs);
  gen_ir_kernel<<<dim3((L * FP + 255) / 256), dim3(256), 0, stream>>>(wir, wii);

  cvt(xp, xp_h, RQ, L, RQ, L);
  cvt(fc1wr, w1r, D, F, D, FP);  cvt(fc1wi, w1i, D, F, D, FP);
  cvt(fc2wr, w2r, D, D, D, D);   cvt(fc2wi, w2i, D, D, D, D);
  cvt(fc3wr, w3r, F, D, FP, D);  cvt(fc3wi, w3i, F, D, FP, D);
  cvt(fc4wr, w4r, D, F, D, FP);  cvt(fc4wi, w4i, D, F, D, FP);
  cvt(fc5wr, w5r, D, D, D, D);   cvt(fc5wi, w5i, D, D, D, D);
  for (int p2 = 0; p2 < 2; ++p2) {
    cvt(AW[p2][0], wAq[p2], D, D, D, D);
    cvt(AW[p2][2], wAk[p2], D, D, D, D);
    cvt(AW[p2][4], wAv[p2], D, D, D, D);
    cvt(AW[p2][6], wAo[p2], D, D, D, D);
  }
  cvt(pw1, wp1, 4 * P, 2 * P, 4 * P, 2 * P);
  cvt(pw2, wp2, P, 4 * P, P, 4 * P);
  pad_bias_kernel<<<dim3(1), dim3(128), 0, stream>>>(fc3br, b3r, F, FP);
  pad_bias_kernel<<<dim3(1), dim3(128), 0, stream>>>(fc3bi, b3i, F, FP);

  // ---- 1) rfft as two real GEMMs: xf = xp @ DFT^T ----
  launch_rr(xp_h, L, wfc, L, nullptr, nullptr, 0, xfr, nullptr, FP, RQ, FP, L, 0);
  launch_rr(xp_h, L, wfs, L, nullptr, nullptr, 0, xfi, nullptr, FP, RQ, FP, L, 0);

  // ---- 2) sigmoid frequency gate, 3) xt = xf x temp ----
  {
    long tot = (long)RQ * FP;
    scale_xf_kernel<<<dim3((unsigned)((tot + 255) / 256)), dim3(256), 0, stream>>>(
        xfr, xfi, fw);
    tot = (long)RKV * FP;
    make_xt_kernel<<<dim3((unsigned)((tot + 255) / 256)), dim3(256), 0, stream>>>(
        xfr, xfi, tmr, tmi, xtr, xti);
  }

  // ---- 4) fc1 (CReLU) -> fc2 : e (B,C,N,M,D) ----
  launch_cc(xtr, xti, FP, w1r, w1i, FP, fc1br, fc1bi, t1r, t1i, D, RKV, D, FP, 1);
  launch_cc(t1r, t1i, D, w2r, w2i, D, fc2br, fc2bi, er, ei, D, RKV, D, D, 0);

  // ---- 5) fc4 (CReLU) -> fc5 : g (B,C,N,D) ----
  launch_cc(xfr, xfi, FP, w4r, w4i, FP, fc4br, fc4bi, t2r, t2i, D, RQ, D, FP, 1);
  launch_cc(t2r, t2i, D, w5r, w5i, D, fc5br, fc5bi, gr, gi, D, RQ, D, D, 0);

  // ---- 6) attention, real part then imag part ----
  const float scale = 0.044194173824159216f;  // D^-0.5
  for (int p2 = 0; p2 < 2; ++p2) {
    const half_t* gp = p2 ? gi : gr;
    const half_t* ep = p2 ? ei : er;
    // Q/K/V projections
    launch_rr(gp, D, wAq[p2], D, AW[p2][1], nullptr, 0, qb, nullptr, D, RQ, D, D, 0);
    launch_rr(ep, D, wAk[p2], D, AW[p2][3], nullptr, 0, kb, nullptr, D, RKV, D, D, 0);
    launch_rr(ep, D, wAv[p2], D, AW[p2][5], nullptr, 0, vb, nullptr, D, RKV, D, D, 0);
    // head transpose of V
    {
      long tot = (long)NBC * H * HD * KV;
      transpose_v_kernel<<<dim3((unsigned)((tot + 255) / 256)), dim3(256), 0,
                           stream>>>(vb, vt);
    }
    // scores[z=bc*H+h] (128 x 1024) = Q_h @ K_h^T   (f32 out)
    launch_rr_b(qb, D, (long)N * D, HD,
                kb, D, (long)KV * D, HD,
                nullptr, sc, KV, (long)H * N * KV, (long)N * KV,
                N, KV, HD, NBC * H, H);
    // softmax over kv
    softmax_kernel<<<dim3((unsigned)(NBC * H * N / 8)), dim3(256), 0, stream>>>(
        sc, pr, NBC * H * N, KV, scale);
    // out_h (128 x 64) = probs @ V_h ; write into head slice of ao
    launch_rr_b(pr, KV, (long)H * N * KV, (long)N * KV,
                vt, KV, (long)H * HD * KV, (long)HD * KV,
                ao, nullptr, D, (long)N * D, HD,
                N, HD, KV, NBC * H, H);
    // output projection + residual g
    launch_rr(ao, D, wAo[p2], D, AW[p2][7], gp, D, outp[p2], nullptr, D,
              RQ, D, D, 0);
  }

  // ---- 7) fc3 (complex) -> o (B,C,N,F) ----
  launch_cc(outp[0], outp[1], D, w3r, w3i, D, b3r, b3i, o_r, o_i, FP,
            RQ, FP, D, 0);

  // ---- 8) irfft as dual real GEMM: ot = o_r@Cr^T + o_i@Ci^T ----
  {
    dim3 g((unsigned)(L / 32), (unsigned)(RQ / 128), 1);
    gemm_rr_kernel<1><<<g, dim3(128), 0, stream>>>(
        o_r, FP, 0, 0, wir, FP, 0, 0, o_i, FP, wii, FP,
        nullptr, nullptr, 0, ot, nullptr, L, 0, 0, RQ, L, FP, 1, 0);
  }

  // ---- 9) projections: relu(ot@W1^T+b1) @ W2^T + b2 -> f32 output ----
  launch_rr(ot, L, wp1, 2 * P, pb1, nullptr, 0, hb, nullptr, 4 * P,
            RQ, 4 * P, L, 1);
  launch_rr(hb, 4 * P, wp2, 4 * P, pb2, nullptr, 0, nullptr, (float*)d_out, P,
            RQ, P, 4 * P, 0);
}
